// CNNSigFF_85633057947684
// MI455X (gfx1250) — matrix operand order repaired and checked by reading, not compiled, not run on hardware
//
#include <hip/hip_runtime.h>
#include <hip/hip_bf16.h>

// ---------------- problem constants (from reference) ----------------
#define NPATH   256     // B*OUT_CH
#define TSTEPS  128     // T'
#define DD      9       // CHANNELS + time
#define S2N     81
#define S3N     729
#define S4N     6561
#define SIGN    7380    // 9+81+729+6561
#define KTOT    29520   // OUT_CH * SIGN
#define KCHUNK  7392    // K-split chunk (mult of 32; last chunk = 7344 = 229*32+16)
#define N0      512
#define N1      256
#define NOUT    10
#define BATCH   64

typedef __attribute__((ext_vector_type(16))) __bf16 v16bf;
typedef __attribute__((ext_vector_type(8)))  float  v8f;

union BFrag { v16bf v; unsigned short u[16]; unsigned w[8]; uint4 q[2]; };

__device__ __forceinline__ unsigned short f2bf(float f) {   // RNE (activations)
    unsigned u = __float_as_uint(f);
    unsigned r = u + 0x7FFFu + ((u >> 16) & 1u);
    return (unsigned short)(r >> 16);
}
// one v_perm_b32: pack {hi16(fhi), hi16(flo)} (bf16 truncation, for weights)
__device__ __forceinline__ unsigned pack2bf(float flo, float fhi) {
    return __builtin_amdgcn_perm(__float_as_uint(fhi), __float_as_uint(flo),
                                 0x07060302u);
}
__device__ __forceinline__ float sigmoidf(float x) {
    return 1.0f / (1.0f + __expf(-x));
}

// =====================================================================
// Kernel 1: fused conv (1x4 stride 4) + time augment + depth-4 signature.
// One block per path; signature state in LDS (32.8 KB).
// Thread map: tid = hi*81 + cd (243 active for S3/S4/E3) makes every LDS
// index "base + compile-time-const" -> ds_load with immediate offsets.
// =====================================================================
__global__ __launch_bounds__(256) void k_conv_sig(
    const float* __restrict__ x,   // [64,1,128,32]
    const float* __restrict__ cw,  // [4,1,1,4]
    const float* __restrict__ cb,  // [4]
    unsigned short* __restrict__ sigB) // [256, 7380] bf16 bits
{
    __shared__ float S[SIGN];   // S1:[0,9) S2:[9,90) S3:[90,819) S4:[819,7380)
    __shared__ float E2s[S2N];
    __shared__ float E3s[S3N];
    __shared__ float dxs[DD];

    const int p   = blockIdx.x;
    const int b   = p >> 2;
    const int oc  = p & 3;
    const int tid = threadIdx.x;

    for (int i = tid; i < SIGN; i += 256) S[i] = 0.0f;

    const float w0 = cw[oc*4+0], w1 = cw[oc*4+1], w2 = cw[oc*4+2], w3 = cw[oc*4+3];
    const float bias = cb[oc];
    float yprev = 0.0f;

    // ---- index precompute (all div/mod hoisted out of the t-loop) ----
    const int cd  = tid % 81;          // (c,d) / (b,c) / (j,k) pair
    const int hi  = tid / 81;          // 0..3 (3 => inactive for 243-maps)
    const int c9  = cd / 9;
    const int d9  = cd % 9;
    const int a0  = 3 * hi;            // 'a' base        (a = a0 + i/9)
    const int ab0 = 27 * hi;           // 'ab' base       (ab = ab0 + i)
    const bool act = (tid < 243);
    const int t9  = tid / 9;           // S2 lane indices (valid tid<81)
    const int t9r = tid % 9;

    __syncthreads();

    float ns4[27], ns3[3], ns2 = 0.0f, ns1 = 0.0f;

    for (int t = 0; t < TSTEPS; ++t) {
        // ---- increment dx (basepoint=True: first increment = path[0]) --
        if (tid < DD) {
            float v;
            if (tid == 0) {
                v = (t == 0) ? 0.0f : (1.0f / 127.0f);       // time channel
            } else {
                const int c = tid - 1;
                const float4 xv = *reinterpret_cast<const float4*>(
                    x + (((size_t)b*TSTEPS + t)*32 + c*4));
                float y = bias + xv.x*w0 + xv.y*w1 + xv.z*w2 + xv.w*w3;
                v = (t == 0) ? y : (y - yprev);
                yprev = y;
            }
            dxs[tid] = v;
        }
        __syncthreads();                                    // (1) dx ready

        // ---- E_k = dx^{⊗k}/k!  (from dx only) -------------------------
        if (tid < S2N) E2s[tid] = dxs[t9] * dxs[t9r] * 0.5f;
        if (act) {
            const float djk = dxs[c9] * dxs[d9] * (1.0f/6.0f);
            #pragma unroll
            for (int j = 0; j < 3; ++j)
                E3s[(a0 + j)*81 + cd] = dxs[a0 + j] * djk;
        }
        __syncthreads();                                    // (2) E ready

        // ---- Chen: S'_k = S_k + E_k + sum_j S_j ⊗ E_{k-j} (old S) -----
        if (act) {
            const float e2cd = E2s[cd];
            const float dxd  = dxs[d9];
            float s1v[3], e3b[9];
            #pragma unroll
            for (int j = 0; j < 3; ++j) s1v[j] = S[a0 + j];
            #pragma unroll
            for (int bq = 0; bq < 9; ++bq) e3b[bq] = E3s[bq*81 + cd];

            #pragma unroll
            for (int i = 0; i < 27; ++i) {       // S4: (a,b,c,d)
                const int ab  = ab0 + i;         // base + const
                const int abc = ab*9 + c9;
                const int idx = ab*81 + cd;
                // E4 + S3⊗dx folded: (S3[abc] + E3[abc]/4) * dx[d]
                const float s3e = S[90 + abc] + 0.25f * E3s[abc];
                ns4[i] = S[819 + idx]
                       + s1v[i/9] * e3b[i%9]     // S1[a] * E3[bcd]
                       + S[9 + ab] * e2cd        // S2[ab] * E2[cd]
                       + s3e * dxd;
            }
            #pragma unroll
            for (int j = 0; j < 3; ++j) {        // S3: (a,b,c)
                const int a   = a0 + j;
                const int idx = a*81 + cd;
                ns3[j] = S[90 + idx] + E3s[idx]
                       + s1v[j] * e2cd           // S1[a] * E2[bc]
                       + S[9 + a*9 + c9] * dxd;  // S2[ab] * dx[c]
            }
        }
        if (tid < S2N) ns2 = S[9 + tid] + E2s[tid] + S[t9] * dxs[t9r];
        if (tid < DD)  ns1 = S[tid] + dxs[tid];
        __syncthreads();                                    // (3) reads done

        if (act) {
            #pragma unroll
            for (int i = 0; i < 27; ++i) S[819 + (ab0 + i)*81 + cd] = ns4[i];
            #pragma unroll
            for (int j = 0; j < 3; ++j)  S[90 + (a0 + j)*81 + cd] = ns3[j];
        }
        if (tid < S2N) S[9 + tid] = ns2;
        if (tid < DD)  S[tid]    = ns1;
        // no barrier here: dx/E writes next iter are fenced by (1)/(2)
    }
    __syncthreads();

    unsigned short* out = sigB + (size_t)p * SIGN;   // flat == [64, 29520]
    for (int i = tid; i < SIGN; i += 256) out[i] = f2bf(S[i]);
}

// =====================================================================
// Kernel: zero fp32 buffer (K-split accumulator; ws is poisoned 0xAA)
// =====================================================================
__global__ void k_zero(float* __restrict__ p, int n) {
    int i = blockIdx.x * blockDim.x + threadIdx.x;
    if (i < n) p[i] = 0.0f;
}

// A fragment (16-bit A-matrix 16x32): per lane two contiguous
// 16B runs at K = k0+kg*16 and K = k0+kg*16+8 (16B aligned by construction).
__device__ __forceinline__ void load_afrag(BFrag& af, const unsigned short* rowA,
                                           int k0, int kg) {
    af.q[0] = *reinterpret_cast<const uint4*>(rowA + k0 + kg*16);
    af.q[1] = *reinterpret_cast<const uint4*>(rowA + k0 + kg*16 + 8);
}

// =====================================================================
// Kernel 2: FC0 via v_wmma_f32_16x16x32_bf16.
// C[64,512] += A_bf16[64,29520] x bf16(W0[29520,512]).
// 4-way K split (branch-free main loop, single half-chunk tail in chunk 3),
// dual N-tiles per wave sharing the A fragment. 256 waves.
// =====================================================================
__global__ __launch_bounds__(256) void k_fc0(
    const unsigned short* __restrict__ A,  // bf16 bits [64, 29520]
    const float* __restrict__ W,           // [29520, 512] fp32
    float* __restrict__ acc)               // [64, 512] zeroed
{
    const int lane = threadIdx.x & 31;
    const int g    = blockIdx.x * 8 + (threadIdx.x >> 5);   // 256 waves
    const int kc   = g & 3;
    const int ntp  = (g >> 2) & 15;
    const int mt   = g >> 6;
    const int kg   = lane >> 4;
    const int ln   = lane & 15;
    const int m    = mt*16 + ln;
    const int nB0  = ntp*32 + ln;
    const int nB1  = nB0 + 16;
    const int kbeg = kc * KCHUNK;
    const int kend = (kbeg + KCHUNK < KTOT) ? (kbeg + KCHUNK) : KTOT;

    const unsigned short* rowA = A + (size_t)m * KTOT;

    v8f c0 = {0.f,0.f,0.f,0.f,0.f,0.f,0.f,0.f};
    v8f c1 = {0.f,0.f,0.f,0.f,0.f,0.f,0.f,0.f};

    int k0 = kbeg;
    for (; k0 + 32 <= kend; k0 += 32) {
        BFrag af, bf0, bf1;
        load_afrag(af, rowA, k0, kg);
        const float* p0 = W + (size_t)(k0 + kg*16) * N0 + nB0;
        #pragma unroll
        for (int v = 0; v < 8; ++v) {           // 1 v_perm_b32 per pair
            bf0.w[v] = pack2bf(p0[(size_t)(2*v)*N0],      p0[(size_t)(2*v+1)*N0]);
            bf1.w[v] = pack2bf(p0[(size_t)(2*v)*N0 + 16], p0[(size_t)(2*v+1)*N0 + 16]);
        }
        c0 = __builtin_amdgcn_wmma_f32_16x16x32_bf16(
                false, af.v, false, bf0.v, (short)0, c0, false, false);
        c1 = __builtin_amdgcn_wmma_f32_16x16x32_bf16(
                false, af.v, false, bf1.v, (short)0, c1, false, false);
    }
    if (k0 < kend) {   // exactly 16 K remain (only kc==3): clean half-chunk
        BFrag af, bf0, bf1;
        af.q[0] = *reinterpret_cast<const uint4*>(rowA + k0);        // K0-7
        af.q[1] = *reinterpret_cast<const uint4*>(rowA + k0 + 8);    // K8-15
        if (kg == 0) {  // lanes 0-15 hold K0-15 (valid); lanes 16-31 -> 0
            const float* p0 = W + (size_t)k0 * N0 + nB0;
            #pragma unroll
            for (int v = 0; v < 8; ++v) {
                bf0.w[v] = pack2bf(p0[(size_t)(2*v)*N0],      p0[(size_t)(2*v+1)*N0]);
                bf1.w[v] = pack2bf(p0[(size_t)(2*v)*N0 + 16], p0[(size_t)(2*v+1)*N0 + 16]);
            }
        } else {
            #pragma unroll
            for (int v = 0; v < 8; ++v) { bf0.w[v] = 0u; bf1.w[v] = 0u; }
        }
        c0 = __builtin_amdgcn_wmma_f32_16x16x32_bf16(
                false, af.v, false, bf0.v, (short)0, c0, false, false);
        c1 = __builtin_amdgcn_wmma_f32_16x16x32_bf16(
                false, af.v, false, bf1.v, (short)0, c1, false, false);
    }

    #pragma unroll
    for (int i = 0; i < 8; ++i) {   // C layout: VGPR i -> M = i + 8*kg, N = lane&15
        int om = mt*16 + i + kg*8;
        atomicAdd(&acc[(size_t)om*N0 + nB0], c0[i]);
        atomicAdd(&acc[(size_t)om*N0 + nB1], c1[i]);
    }
}

// bias + sigmoid, fp32 -> bf16 activations for FC1
__global__ void k_bias_sig0(const float* __restrict__ acc,
                            const float* __restrict__ b0,
                            unsigned short* __restrict__ z0b) {
    int i = blockIdx.x * blockDim.x + threadIdx.x;
    if (i < BATCH * N0) z0b[i] = f2bf(sigmoidf(acc[i] + b0[i & (N0-1)]));
}

// =====================================================================
// Kernel 3: FC1 via WMMA bf16, fused bias+sigmoid epilogue. 64 waves.
// =====================================================================
__global__ __launch_bounds__(256) void k_fc1(
    const unsigned short* __restrict__ A,  // z0b bf16 [64, 512]
    const float* __restrict__ W,           // [512, 256]
    const float* __restrict__ b1,
    float* __restrict__ z1)                // [64, 256]
{
    const int lane = threadIdx.x & 31;
    const int g    = blockIdx.x * 8 + (threadIdx.x >> 5);
    const int nt   = g & 15;
    const int mt   = g >> 4;
    const int kg   = lane >> 4;
    const int ln   = lane & 15;
    const int m    = mt*16 + ln;
    const int nB   = nt*16 + ln;
    const unsigned short* rowA = A + (size_t)m * N0;

    v8f c = {0.f,0.f,0.f,0.f,0.f,0.f,0.f,0.f};

    for (int k0 = 0; k0 < N0; k0 += 32) {
        BFrag af, bf;
        load_afrag(af, rowA, k0, kg);
        const float* p0 = W + (size_t)(k0 + kg*16) * N1 + nB;
        #pragma unroll
        for (int v = 0; v < 8; ++v)
            bf.w[v] = pack2bf(p0[(size_t)(2*v)*N1], p0[(size_t)(2*v+1)*N1]);
        c = __builtin_amdgcn_wmma_f32_16x16x32_bf16(
                false, af.v, false, bf.v, (short)0, c, false, false);
    }

    #pragma unroll
    for (int i = 0; i < 8; ++i) {
        int om = mt*16 + i + kg*8;
        z1[(size_t)om*N1 + nB] = sigmoidf(c[i] + b1[nB]);
    }
}

// =====================================================================
// Kernel 4: FC2 + log_softmax (64x10, negligible; one thread/row)
// =====================================================================
__global__ void k_fc2_lsm(const float* __restrict__ z1,
                          const float* __restrict__ W,   // [256,10]
                          const float* __restrict__ b2,
                          float* __restrict__ out) {     // [64,10]
    int row = blockIdx.x * blockDim.x + threadIdx.x;
    if (row >= BATCH) return;
    float lg[NOUT];
    #pragma unroll
    for (int j = 0; j < NOUT; ++j) lg[j] = b2[j];
    for (int k = 0; k < N1; ++k) {
        float zv = z1[(size_t)row*N1 + k];
        #pragma unroll
        for (int j = 0; j < NOUT; ++j) lg[j] += zv * W[k*NOUT + j];
    }
    float mx = lg[0];
    #pragma unroll
    for (int j = 1; j < NOUT; ++j) mx = fmaxf(mx, lg[j]);
    float s = 0.0f;
    #pragma unroll
    for (int j = 0; j < NOUT; ++j) s += __expf(lg[j] - mx);
    float lse = mx + __logf(s);
    #pragma unroll
    for (int j = 0; j < NOUT; ++j) out[row*NOUT + j] = lg[j] - lse;
}

// =====================================================================
extern "C" void kernel_launch(void* const* d_in, const int* in_sizes, int n_in,
                              void* d_out, int out_size, void* d_ws, size_t ws_size,
                              hipStream_t stream) {
    const float* x  = (const float*)d_in[0];
    const float* cw = (const float*)d_in[1];
    const float* cb = (const float*)d_in[2];
    const float* w0 = (const float*)d_in[3];
    const float* b0 = (const float*)d_in[4];
    const float* w1 = (const float*)d_in[5];
    const float* b1 = (const float*)d_in[6];
    const float* w2 = (const float*)d_in[7];
    const float* b2 = (const float*)d_in[8];
    float* out = (float*)d_out;

    // workspace layout (≈4.0 MB total)
    char* ws = (char*)d_ws;
    unsigned short* sigB = (unsigned short*)ws;                       // 3,778,560 B
    float*          acc0 = (float*)(ws + 3778560);                    //   131,072 B
    unsigned short* z0b  = (unsigned short*)(ws + 3778560 + 131072);  //    65,536 B
    float*          z1   = (float*)(ws + 3778560 + 131072 + 65536);   //    65,536 B

    k_conv_sig<<<NPATH, 256, 0, stream>>>(x, cw, cb, sigB);
    k_zero<<<(BATCH*N0 + 255)/256, 256, 0, stream>>>(acc0, BATCH*N0);
    k_fc0<<<32, 256, 0, stream>>>(sigB, w0, acc0);                    // 256 waves
    k_bias_sig0<<<(BATCH*N0 + 255)/256, 256, 0, stream>>>(acc0, b0, z0b);
    k_fc1<<<8, 256, 0, stream>>>(z0b, w1, b1, z1);                    // 64 waves
    k_fc2_lsm<<<2, 32, 0, stream>>>(z1, w2, b2, out);
    (void)in_sizes; (void)n_in; (void)out_size; (void)ws_size;
}
